// EdgeSelector_63075889709559
// MI455X (gfx1250) — matrix operand order repaired and checked by reading, not compiled, not run on hardware
//
#include <hip/hip_runtime.h>
#include <hip/hip_bf16.h>

// EdgeSelector fused kernel for MI455X (gfx1250), wave32 + WMMA bf16.
// B=32, N=64, D=64, H=4, E=256, HD=64.

typedef __attribute__((ext_vector_type(16))) __bf16 v16bf;
typedef __attribute__((ext_vector_type(8)))  __bf16 v8bf;
typedef __attribute__((ext_vector_type(8)))  float  v8f;

#define EPS 1e-10f

static __device__ __forceinline__ float wave_red_add(float v) {
#pragma unroll
  for (int off = 16; off > 0; off >>= 1) v += __shfl_xor(v, off, 32);
  return v;
}

// ---- WMMA fragment loaders (CDNA5 ISA layouts) ------------------------------
// A-matrix 16x32 bf16: lane (nx = lane&15) holds row m; kh = lane>>4.
//   element e<8 : K = 8*kh + e        (contiguous 16B)
//   element e>=8: K = 16 + 8*kh + (e-8)  (contiguous 16B)
static __device__ __forceinline__ v16bf ld_fragA(const __bf16* base, int row0,
                                                 int col0, int ld) {
  const int lane = threadIdx.x & 31;
  const int nx = lane & 15, kh = lane >> 4;
  const __bf16* p = base + (size_t)(row0 + nx) * ld + col0 + 8 * kh;
  v8bf lo = *(const v8bf*)(p);
  v8bf hi = *(const v8bf*)(p + 16);
  v16bf r;
#pragma unroll
  for (int e = 0; e < 8; ++e) { r[e] = lo[e]; r[e + 8] = hi[e]; }
  return r;
}

// B-matrix 32x16 bf16 read from N-major storage (row = output col n, stride=K):
//   lane nx holds column n; element e<8: K = 8*kh + e; e>=8: K = 16 + 8*kh + (e-8).
static __device__ __forceinline__ v16bf ld_fragB(const __bf16* base, int nrow0,
                                                 int kcol0, int ld) {
  const int lane = threadIdx.x & 31;
  const int nx = lane & 15, kh = lane >> 4;
  const __bf16* p = base + (size_t)(nrow0 + nx) * ld + kcol0 + 8 * kh;
  v8bf lo = *(const v8bf*)(p);
  v8bf hi = *(const v8bf*)(p + 16);
  v16bf r;
#pragma unroll
  for (int e = 0; e < 8; ++e) { r[e] = lo[e]; r[e + 8] = hi[e]; }
  return r;
}

static __device__ __forceinline__ v8f wmma_bf16(v16bf a, v16bf b, v8f c) {
  return __builtin_amdgcn_wmma_f32_16x16x32_bf16(false, a, false, b,
                                                 (short)0, c, false, false);
}

// ---- Prep kernels -----------------------------------------------------------
__global__ void k_cvt_bf16(const float* __restrict__ src,
                           __bf16* __restrict__ dst, int n) {
  int i = blockIdx.x * blockDim.x + threadIdx.x;
  if (i < n) dst[i] = (__bf16)src[i];
}

// One wave per (b,node): LayerNorm(x) * ped, emitted as bf16.
__global__ void k_node_ln(const float* __restrict__ x,
                          const int* __restrict__ mask,
                          const float* __restrict__ g,
                          const float* __restrict__ bb,
                          __bf16* __restrict__ xn) {
  const int row = blockIdx.x;       // b*64 + node
  const int lane = threadIdx.x;     // 0..31
  float x0 = x[row * 64 + lane], x1 = x[row * 64 + lane + 32];
  float m = wave_red_add(x0 + x1) * (1.f / 64.f);
  float d0 = x0 - m, d1 = x1 - m;
  float var = wave_red_add(d0 * d0 + d1 * d1) * (1.f / 64.f);
  float rs = rsqrtf(var + 1e-5f);
  float msum = wave_red_add((float)(mask[row * 64 + lane] +
                                    mask[row * 64 + lane + 32]));
  float ped = (msum > 0.f) ? 1.f : 0.f;
  xn[row * 64 + lane]      = (__bf16)((d0 * rs * g[lane] + bb[lane]) * ped);
  xn[row * 64 + lane + 32] = (__bf16)((d1 * rs * g[lane + 32] + bb[lane + 32]) * ped);
}

// ---- Main fused kernel: one workgroup per (b, target) -----------------------
__global__ __launch_bounds__(256) void k_edge_main(
    const float* __restrict__ A, const int* __restrict__ attn_mask,
    const float* __restrict__ gumbel_u,
    const float* __restrict__ leg, const float* __restrict__ leb,
    const float* __restrict__ in_b, const float* __restrict__ b1,
    const float* __restrict__ w2, const float* __restrict__ b2,
    const __bf16* __restrict__ wq_b,   // 768x256 bf16 (N-major == torch row-major)
    const __bf16* __restrict__ w1_b,   // 64x64 bf16 (N-major == row-major (o,d))
    const __bf16* __restrict__ xn_b,   // B*64 x 64 bf16
    float* __restrict__ out_em, float* __restrict__ out_sm) {
  extern __shared__ char smem[];
  __bf16* s_ap = (__bf16*)smem;                  // 64x256 bf16 (32KB), reused as A2
  __bf16* s_q  = (__bf16*)(smem + 32768);        // 64x256 bf16 (q, pre-scaled)
  __bf16* s_k  = (__bf16*)(smem + 65536);        // 64x256 bf16
  __bf16* s_vt = (__bf16*)(smem + 98304);        // per head h: v^T, d-major 64x64
  float*  s_sc = (float*)(smem + 131072);        // 4x64x64 f32 (64KB), reused as hmid
  __bf16* s_w  = (__bf16*)(smem + 196608);       // 4x64x64 bf16 attn weights
  float*  s_ms = (float*)(smem + 229376);        // misc (8KB)
  float* s_mask = s_ms;          // 64
  float* s_b1   = s_ms + 64;     // 64
  float* s_w2   = s_ms + 128;    // 64
  float* s_b2   = s_ms + 192;    // 1
  float* s_lg   = s_ms + 200;    // 256
  float* s_em   = s_ms + 456;    // 256
  float* s_z    = s_ms + 712;    // 256
  float* s_red  = s_ms + 968;    // 8
  float* s_leg  = s_ms + 976;    // 128
  float* s_leb  = s_ms + 1104;   // 128
  __bf16* s_a2 = s_ap;
  float*  s_hm = s_sc;

  const int tid = threadIdx.x, lane = tid & 31, wave = tid >> 5;
  const int nx = lane & 15, kh = lane >> 4;
  const int bt = blockIdx.x, b = bt >> 6, t = bt & 63;

  // -------- preload small constants --------
  if (tid < 64) {
    s_mask[tid] = (float)attn_mask[(size_t)bt * 64 + tid];  // mask[b,t,j]
    s_b1[tid] = b1[tid];
    s_w2[tid] = w2[tid];
  }
  if (tid < 128) { s_leg[tid] = leg[tid]; s_leb[tid] = leb[tid]; }
  if (tid == 128) s_b2[0] = b2[0];
  __syncthreads();

  // -------- Phase A: build Ap row j = [xn_j | xn_t | LN(A[b,j,t])*mask[t,j]] --------
  const __bf16* xnb = xn_b + (size_t)b * 64 * 64;
  const __bf16* xnt = xnb + (size_t)t * 64;
#pragma unroll 1
  for (int rr = 0; rr < 8; ++rr) {
    const int j = wave * 8 + rr;
    const float* ar = A + (((size_t)b * 64 + j) * 64 + t) * 128;
    float v0 = ar[lane], v1 = ar[lane + 32], v2 = ar[lane + 64], v3 = ar[lane + 96];
    float m = wave_red_add(v0 + v1 + v2 + v3) * (1.f / 128.f);
    float d0 = v0 - m, d1 = v1 - m, d2 = v2 - m, d3 = v3 - m;
    float var = wave_red_add(d0 * d0 + d1 * d1 + d2 * d2 + d3 * d3) * (1.f / 128.f);
    float rs = rsqrtf(var + 1e-5f);
    float mj = s_mask[j];
    __bf16* row = s_ap + j * 256;
    row[128 + lane]      = (__bf16)((d0 * rs * s_leg[lane]      + s_leb[lane])      * mj);
    row[128 + lane + 32] = (__bf16)((d1 * rs * s_leg[lane + 32] + s_leb[lane + 32]) * mj);
    row[128 + lane + 64] = (__bf16)((d2 * rs * s_leg[lane + 64] + s_leb[lane + 64]) * mj);
    row[128 + lane + 96] = (__bf16)((d3 * rs * s_leg[lane + 96] + s_leb[lane + 96]) * mj);
    const __bf16* xnj = xnb + (size_t)j * 64;
    row[lane]       = xnj[lane];
    row[lane + 32]  = xnj[lane + 32];
    row[64 + lane]      = xnt[lane];
    row[64 + lane + 32] = xnt[lane + 32];
  }
  __syncthreads();

  // -------- Phase B: qkv = Ap(64x256) @ W^T(256x768), bf16 WMMA ----------------
#pragma unroll 1
  for (int ct = 0; ct < 6; ++ct) {
    const int col0 = (wave * 6 + ct) * 16;
    v8f acc[4];
#pragma unroll
    for (int rt = 0; rt < 4; ++rt) {
      v8f z = {0.f, 0.f, 0.f, 0.f, 0.f, 0.f, 0.f, 0.f};
      acc[rt] = z;
    }
#pragma unroll
    for (int ks = 0; ks < 8; ++ks) {
      v16bf bf = ld_fragB(wq_b, col0, ks * 32, 256);   // global (L2-resident weights)
#pragma unroll
      for (int rt = 0; rt < 4; ++rt)
        acc[rt] = wmma_bf16(ld_fragA(s_ap, rt * 16, ks * 32, 256), bf, acc[rt]);
    }
    const int f = col0 + nx;
    const float bias = in_b[f];
#pragma unroll
    for (int rt = 0; rt < 4; ++rt) {
#pragma unroll
      for (int r = 0; r < 8; ++r) {
        const int mrow = rt * 16 + r + 8 * kh;
        const float val = acc[rt][r] + bias;
        if (f < 256) {                                   // q, pre-scaled by hd^-0.5
          s_q[mrow * 256 + f] = (__bf16)(val * 0.125f);
        } else if (f < 512) {                            // k
          s_k[mrow * 256 + (f - 256)] = (__bf16)val;
        } else {                                         // v, stored transposed
          const int fv = f - 512, h = fv >> 6, d = fv & 63;
          s_vt[h * 4096 + d * 64 + mrow] = (__bf16)val;
        }
      }
    }
  }
  __syncthreads();

  // -------- Phase C: scores[h] = q_h(64x64) @ k_h^T --------
  const int h = wave >> 1, rhalf = wave & 1;
#pragma unroll 1
  for (int rt = 0; rt < 2; ++rt) {
    const int row0 = (rhalf * 2 + rt) * 16;
#pragma unroll 1
    for (int ct = 0; ct < 4; ++ct) {
      v8f acc = {0.f, 0.f, 0.f, 0.f, 0.f, 0.f, 0.f, 0.f};
#pragma unroll
      for (int ks = 0; ks < 2; ++ks) {
        v16bf af = ld_fragA(s_q, row0, h * 64 + ks * 32, 256);
        v16bf bf = ld_fragB(s_k, ct * 16, h * 64 + ks * 32, 256);
        acc = wmma_bf16(af, bf, acc);
      }
#pragma unroll
      for (int r = 0; r < 8; ++r)
        s_sc[h * 4096 + (row0 + r + 8 * kh) * 64 + ct * 16 + nx] = acc[r];
    }
  }
  __syncthreads();

  // -------- Phase C2: masked softmax + renorm; one thread per (head,row) -------
  {
    const int hh = tid >> 6, j = tid & 63;
    float* row = s_sc + hh * 4096 + j * 64;
    float mx = -3.4e38f;
    for (int k2 = 0; k2 < 64; ++k2) mx = fmaxf(mx, row[k2]);
    float Z = 0.f, S = 0.f;
    for (int k2 = 0; k2 < 64; ++k2) {
      float e = __expf(row[k2] - mx);
      row[k2] = e; Z += e; S += e * s_mask[k2];
    }
    // w = softmax*mj*mk / (sum + EPS)  ==  e*mj*mk / (mj*S + EPS*Z)
    const float mj = s_mask[j];
    const float inv = 1.f / (mj * S + EPS * Z);
    __bf16* wr = s_w + hh * 4096 + j * 64;
    for (int k2 = 0; k2 < 64; ++k2)
      wr[k2] = (__bf16)(mj * s_mask[k2] * row[k2] * inv);
  }
  __syncthreads();

  // -------- Phase D: A2[h] = w_h(64x64) @ v_h(64x64), v stored d-major ---------
#pragma unroll 1
  for (int rt = 0; rt < 2; ++rt) {
    const int row0 = (rhalf * 2 + rt) * 16;
#pragma unroll 1
    for (int ct = 0; ct < 4; ++ct) {
      v8f acc = {0.f, 0.f, 0.f, 0.f, 0.f, 0.f, 0.f, 0.f};
#pragma unroll
      for (int ks = 0; ks < 2; ++ks) {
        v16bf af = ld_fragA(s_w + h * 4096, row0, ks * 32, 64);
        v16bf bf = ld_fragB(s_vt + h * 4096, ct * 16, ks * 32, 64);
        acc = wmma_bf16(af, bf, acc);
      }
#pragma unroll
      for (int r = 0; r < 8; ++r)
        s_a2[h * 4096 + (row0 + r + 8 * kh) * 64 + ct * 16 + nx] = (__bf16)acc[r];
    }
  }
  __syncthreads();

  // -------- Phase E: hmid = relu(A2 @ w1^T + b1) --------
#pragma unroll 1
  for (int rt = 0; rt < 2; ++rt) {
    const int row0 = (rhalf * 2 + rt) * 16;
#pragma unroll 1
    for (int ct = 0; ct < 4; ++ct) {
      v8f acc = {0.f, 0.f, 0.f, 0.f, 0.f, 0.f, 0.f, 0.f};
#pragma unroll
      for (int ks = 0; ks < 2; ++ks) {
        v16bf af = ld_fragA(s_a2 + h * 4096, row0, ks * 32, 64);
        v16bf bf = ld_fragB(w1_b, ct * 16, ks * 32, 64);
        acc = wmma_bf16(af, bf, acc);
      }
#pragma unroll
      for (int r = 0; r < 8; ++r) {
        const int o = ct * 16 + nx;
        s_hm[h * 4096 + (row0 + r + 8 * kh) * 64 + o] = fmaxf(acc[r] + s_b1[o], 0.f);
      }
    }
  }
  __syncthreads();

  // -------- Phase F: logits, em, gumbel-softmax --------
  {
    const int hh = tid >> 6, j = tid & 63;
    const float* hr = s_hm + hh * 4096 + j * 64;
    float lg = s_b2[0];
    for (int o = 0; o < 64; ++o) lg += hr[o] * s_w2[o];
    s_lg[tid] = lg;
  }
  __syncthreads();
  if (tid < 4) {  // per-head softmax over neighbors + mask renorm
    const float* row = s_lg + tid * 64;
    float mx = -3.4e38f;
    for (int j2 = 0; j2 < 64; ++j2) mx = fmaxf(mx, row[j2]);
    float Z = 0.f, S = 0.f;
    float* er = s_em + tid * 64;
    for (int j2 = 0; j2 < 64; ++j2) {
      float e = __expf(row[j2] - mx);
      er[j2] = e; Z += e; S += e * s_mask[j2];
    }
    const float inv = 1.f / (S + EPS * Z);     // em = e*mj / (S + EPS*Z)
    for (int j2 = 0; j2 < 64; ++j2) er[j2] = er[j2] * s_mask[j2] * inv;
  }
  __syncthreads();
  {
    const int hh = tid >> 6, j = tid & 63;
    const float em = s_em[tid];
    const size_t oidx = ((size_t)bt * 4 + hh) * 64 + j;
    out_em[oidx] = em;
    const float u = gumbel_u[oidx];
    const float g = -__logf(-__logf(u + EPS) + EPS);
    s_z[tid] = __logf(em + EPS) + g;           // TAU == 1
  }
  __syncthreads();
  if (tid < 4) {
    const float* row = s_z + tid * 64;
    float mx = -3.4e38f;
    for (int j2 = 0; j2 < 64; ++j2) mx = fmaxf(mx, row[j2]);
    float Z = 0.f;
    for (int j2 = 0; j2 < 64; ++j2) Z += __expf(row[j2] - mx);
    s_red[tid * 2] = mx; s_red[tid * 2 + 1] = Z;
  }
  __syncthreads();
  {
    const int hh = tid >> 6;
    const float sm = __expf(s_z[tid] - s_red[hh * 2]) / s_red[hh * 2 + 1];
    out_sm[((size_t)bt * 4 + hh) * 64 + (tid & 63)] = sm;
  }
}

// ---- Host launcher ----------------------------------------------------------
extern "C" void kernel_launch(void* const* d_in, const int* in_sizes, int n_in,
                              void* d_out, int out_size, void* d_ws, size_t ws_size,
                              hipStream_t stream) {
  (void)in_sizes; (void)n_in; (void)out_size; (void)ws_size;
  const float* x    = (const float*)d_in[0];
  const float* A    = (const float*)d_in[1];
  const int*   am   = (const int*)d_in[2];
  const float* gu   = (const float*)d_in[3];
  const float* lng  = (const float*)d_in[4];
  const float* lnb  = (const float*)d_in[5];
  const float* leg  = (const float*)d_in[6];
  const float* leb  = (const float*)d_in[7];
  const float* wq   = (const float*)d_in[8];
  const float* inb  = (const float*)d_in[9];
  const float* w1   = (const float*)d_in[10];
  const float* b1   = (const float*)d_in[11];
  const float* w2   = (const float*)d_in[12];
  const float* b2   = (const float*)d_in[13];

  char* ws = (char*)d_ws;
  __bf16* wq_b = (__bf16*)ws;                              // 768*256 bf16
  __bf16* w1_b = (__bf16*)(ws + 196608 * 2);               // 64*64 bf16
  __bf16* xn_b = (__bf16*)(ws + 196608 * 2 + 4096 * 2);    // 2048*64 bf16

  k_cvt_bf16<<<768, 256, 0, stream>>>(wq, wq_b, 196608);
  k_cvt_bf16<<<16, 256, 0, stream>>>(w1, w1_b, 4096);
  k_node_ln<<<2048, 32, 0, stream>>>(x, am, lng, lnb, xn_b);

  float* out_em = (float*)d_out;
  float* out_sm = out_em + (size_t)32 * 64 * 4 * 64;
  const size_t smem = 229376 + 2048 * 4;  // 237,568 B (< 320 KB WGP LDS)
  k_edge_main<<<2048, 256, smem, stream>>>(A, am, gu, leg, leb, inb, b1, w2, b2,
                                           wq_b, w1_b, xn_b, out_em, out_sm);
}
